// GAT_53266184405050
// MI455X (gfx1250) — compile-verified, matrix operand order checked by reading, not run on hardware
//
#include <hip/hip_runtime.h>
#include <hip/hip_bf16.h>
#include <math.h>

typedef __attribute__((ext_vector_type(16))) _Float16 v16h;
typedef __attribute__((ext_vector_type(8)))  _Float16 v8h;
typedef __attribute__((ext_vector_type(8)))  float    v8f;

#define NNODES 100000
#define NEDGES 1600000
#define FIN    256
#define FOUT   128

// ---- float <-> order-preserving uint (for atomicMax over signed floats) ----
__device__ __forceinline__ unsigned f2ord(float f) {
    unsigned u = __float_as_uint(f);
    return (u & 0x80000000u) ? ~u : (u | 0x80000000u);
}
__device__ __forceinline__ float ord2f(unsigned u) {
    return __uint_as_float((u & 0x80000000u) ? (u ^ 0x80000000u) : ~u);
}

// ---- 0) W [256,128] f32 -> Wt [128][256] f16 (transposed, contiguous cols) ----
__global__ void k_convW(const float* __restrict__ W, _Float16* __restrict__ Wt) {
    int t = blockIdx.x * blockDim.x + threadIdx.x;
    if (t < FIN * FOUT) {
        int n = t / FIN, k = t % FIN;
        Wt[n * FIN + k] = (_Float16)W[k * FOUT + n];
    }
}

// ---- 1) WMMA GEMM: H[N,128] = in_feat[N,256] @ W ----
// One wave32 per 16x16 output tile; block = 8 waves = 8 column tiles of one row tile.
__global__ void __launch_bounds__(256) k_gemm(const float* __restrict__ A,
                                              const _Float16* __restrict__ Wt,
                                              float* __restrict__ H) {
    const int lane  = threadIdx.x & 31;
    const int wave  = threadIdx.x >> 5;     // ntile 0..7
    const int half  = lane >> 4;            // K-group selector (ISA 16-bit A/B layout)
    const int mcol  = lane & 15;            // M index (A) / N index (B)
    const int mtile = blockIdx.x;           // 0..6249  (100000/16 exact)

    const int arow = mtile * 16 + mcol;
    const int bcol = wave  * 16 + mcol;

    const float4* __restrict__ Ar = reinterpret_cast<const float4*>(A + (size_t)arow * FIN);
    const v8h*    __restrict__ Bc = reinterpret_cast<const v8h*>(Wt + (size_t)bcol * FIN);

    v8f c = {};
#pragma unroll
    for (int kk = 0; kk < FIN / 32; ++kk) {
        // A: lane holds K = kbase + 8*half + [0..7] and kbase + 16 + 8*half + [0..7]
        const float4 a0 = Ar[kk * 8 + 2 * half + 0];
        const float4 a1 = Ar[kk * 8 + 2 * half + 1];
        const float4 a2 = Ar[kk * 8 + 4 + 2 * half + 0];
        const float4 a3 = Ar[kk * 8 + 4 + 2 * half + 1];
        v16h a;
        a[0]=(_Float16)a0.x;  a[1]=(_Float16)a0.y;  a[2]=(_Float16)a0.z;  a[3]=(_Float16)a0.w;
        a[4]=(_Float16)a1.x;  a[5]=(_Float16)a1.y;  a[6]=(_Float16)a1.z;  a[7]=(_Float16)a1.w;
        a[8]=(_Float16)a2.x;  a[9]=(_Float16)a2.y;  a[10]=(_Float16)a2.z; a[11]=(_Float16)a2.w;
        a[12]=(_Float16)a3.x; a[13]=(_Float16)a3.y; a[14]=(_Float16)a3.z; a[15]=(_Float16)a3.w;

        // B (pre-transposed f16): same K striping, contiguous 16B runs
        const v8h b0 = Bc[kk * 4 + half];
        const v8h b1 = Bc[kk * 4 + 2 + half];
        v16h b;
#pragma unroll
        for (int j = 0; j < 8; ++j) { b[j] = b0[j]; b[8 + j] = b1[j]; }

        c = __builtin_amdgcn_wmma_f32_16x16x32_f16(false, a, false, b, (short)0, c, false, false);
    }

    // C/D layout: VGPR r -> M = r + 8*half, N = mcol
    float* Hp = H + (size_t)(mtile * 16 + half * 8) * FOUT + wave * 16 + mcol;
#pragma unroll
    for (int r = 0; r < 8; ++r) Hp[(size_t)r * FOUT] = c[r];
}

// ---- 2) el/er row reductions: one wave per node ----
__global__ void __launch_bounds__(256) k_rowred(const float* __restrict__ H,
                                                const float* __restrict__ al,
                                                const float* __restrict__ ar,
                                                float* __restrict__ el,
                                                float* __restrict__ er) {
    int node = blockIdx.x * 8 + (threadIdx.x >> 5);
    int lane = threadIdx.x & 31;
    if (node >= NNODES) return;
    float4 hv = reinterpret_cast<const float4*>(H + (size_t)node * FOUT)[lane];
    float4 lv = reinterpret_cast<const float4*>(al)[lane];
    float4 rv = reinterpret_cast<const float4*>(ar)[lane];
    float sl = hv.x * lv.x + hv.y * lv.y + hv.z * lv.z + hv.w * lv.w;
    float sr = hv.x * rv.x + hv.y * rv.y + hv.z * rv.z + hv.w * rv.w;
#pragma unroll
    for (int off = 16; off; off >>= 1) {
        sl += __shfl_xor(sl, off, 32);
        sr += __shfl_xor(sr, off, 32);
    }
    if (lane == 0) { el[node] = sl; er[node] = sr; }
}

// ---- 3) init agg=0, s=0, m=-inf(ordered) ----
__global__ void k_init(float* __restrict__ agg, float* __restrict__ s,
                       unsigned* __restrict__ mbuf) {
    int t = blockIdx.x * blockDim.x + threadIdx.x;
    if (t < NNODES * FOUT) agg[t] = 0.0f;
    if (t < NNODES) { s[t] = 0.0f; mbuf[t] = f2ord(-INFINITY); }
}

// ---- 4) segment max over edges ----
__global__ void k_edge_max(const int* __restrict__ src, const int* __restrict__ dst,
                           const float* __restrict__ el, const float* __restrict__ er,
                           unsigned* __restrict__ mbuf) {
    int t = blockIdx.x * blockDim.x + threadIdx.x;
    if (t >= NEDGES) return;
    int d = dst[t];
    float x = el[src[t]] + er[d];
    float e = x > 0.0f ? x : 0.2f * x;
    atomicMax(&mbuf[d], f2ord(e));
}

// ---- 5) segment sum of exp(e - m) ----
__global__ void k_edge_sum(const int* __restrict__ src, const int* __restrict__ dst,
                           const float* __restrict__ el, const float* __restrict__ er,
                           const unsigned* __restrict__ mbuf, float* __restrict__ s) {
    int t = blockIdx.x * blockDim.x + threadIdx.x;
    if (t >= NEDGES) return;
    int d = dst[t];
    float x = el[src[t]] + er[d];
    float e = x > 0.0f ? x : 0.2f * x;
    atomicAdd(&s[d], __expf(e - ord2f(mbuf[d])));
}

// ---- 6) attention-weighted aggregation: one wave per edge ----
__global__ void __launch_bounds__(256) k_edge_agg(const int* __restrict__ src,
                                                  const int* __restrict__ dst,
                                                  const float* __restrict__ el,
                                                  const float* __restrict__ er,
                                                  const unsigned* __restrict__ mbuf,
                                                  const float* __restrict__ s,
                                                  const float* __restrict__ H,
                                                  float* __restrict__ agg) {
    int edge = blockIdx.x * 8 + (threadIdx.x >> 5);
    int lane = threadIdx.x & 31;
    if (edge >= NEDGES) return;
    int sn = src[edge], dn = dst[edge];
    float x = el[sn] + er[dn];
    float e = x > 0.0f ? x : 0.2f * x;
    float sv = s[dn];
    sv = (sv == 0.0f) ? 1.0f : sv;
    float alpha = __expf(e - ord2f(mbuf[dn])) / sv;
    float4 hv = reinterpret_cast<const float4*>(H + (size_t)sn * FOUT)[lane];
    float* ap = agg + (size_t)dn * FOUT + lane * 4;
    atomicAdd(ap + 0, alpha * hv.x);
    atomicAdd(ap + 1, alpha * hv.y);
    atomicAdd(ap + 2, alpha * hv.z);
    atomicAdd(ap + 3, alpha * hv.w);
}

// ---- 7) head: bias -> relu -> [128x2] linear -> sigmoid; one wave per node ----
__global__ void __launch_bounds__(256) k_head(const float* __restrict__ agg,
                                              const float* __restrict__ bias,
                                              const float* __restrict__ fcw,
                                              const float* __restrict__ fcb,
                                              float* __restrict__ out) {
    int node = blockIdx.x * 8 + (threadIdx.x >> 5);
    int lane = threadIdx.x & 31;
    if (node >= NNODES) return;
    float4 av = reinterpret_cast<const float4*>(agg + (size_t)node * FOUT)[lane];
    float4 bv = reinterpret_cast<const float4*>(bias)[lane];
    int c = lane * 4;
    float t0 = 0.0f, t1 = 0.0f, t;
    t = fmaxf(av.x + bv.x, 0.0f); t0 += t * fcw[(c + 0) * 2 + 0]; t1 += t * fcw[(c + 0) * 2 + 1];
    t = fmaxf(av.y + bv.y, 0.0f); t0 += t * fcw[(c + 1) * 2 + 0]; t1 += t * fcw[(c + 1) * 2 + 1];
    t = fmaxf(av.z + bv.z, 0.0f); t0 += t * fcw[(c + 2) * 2 + 0]; t1 += t * fcw[(c + 2) * 2 + 1];
    t = fmaxf(av.w + bv.w, 0.0f); t0 += t * fcw[(c + 3) * 2 + 0]; t1 += t * fcw[(c + 3) * 2 + 1];
#pragma unroll
    for (int off = 16; off; off >>= 1) {
        t0 += __shfl_xor(t0, off, 32);
        t1 += __shfl_xor(t1, off, 32);
    }
    if (lane == 0) {
        out[(size_t)node * 2 + 0] = 1.0f / (1.0f + __expf(-(t0 + fcb[0])));
        out[(size_t)node * 2 + 1] = 1.0f / (1.0f + __expf(-(t1 + fcb[1])));
    }
}

extern "C" void kernel_launch(void* const* d_in, const int* in_sizes, int n_in,
                              void* d_out, int out_size, void* d_ws, size_t ws_size,
                              hipStream_t stream) {
    const float* in_feat = (const float*)d_in[0];
    const float* W       = (const float*)d_in[1];
    const float* attn_l  = (const float*)d_in[2];
    const float* attn_r  = (const float*)d_in[3];
    const float* bias    = (const float*)d_in[4];
    const float* fc_w    = (const float*)d_in[5];
    const float* fc_b    = (const float*)d_in[6];
    const int*   src     = (const int*)d_in[7];
    const int*   dst     = (const int*)d_in[8];
    float*       out     = (float*)d_out;

    // workspace layout (all offsets 16B-aligned)
    char* ws = (char*)d_ws;
    size_t off = 0;
    float* H   = (float*)(ws + off);    off += (size_t)NNODES * FOUT * sizeof(float);
    float* agg = (float*)(ws + off);    off += (size_t)NNODES * FOUT * sizeof(float);
    float* el  = (float*)(ws + off);    off += (size_t)NNODES * sizeof(float);
    float* er  = (float*)(ws + off);    off += (size_t)NNODES * sizeof(float);
    float* s   = (float*)(ws + off);    off += (size_t)NNODES * sizeof(float);
    unsigned* mbuf = (unsigned*)(ws + off); off += (size_t)NNODES * sizeof(unsigned);
    _Float16* Wt   = (_Float16*)(ws + off);

    k_convW<<<(FIN * FOUT + 255) / 256, 256, 0, stream>>>(W, Wt);
    k_gemm<<<NNODES / 16, 256, 0, stream>>>(in_feat, Wt, H);
    k_rowred<<<(NNODES + 7) / 8, 256, 0, stream>>>(H, attn_l, attn_r, el, er);
    k_init<<<(NNODES * FOUT + 255) / 256, 256, 0, stream>>>(agg, s, mbuf);
    k_edge_max<<<(NEDGES + 255) / 256, 256, 0, stream>>>(src, dst, el, er, mbuf);
    k_edge_sum<<<(NEDGES + 255) / 256, 256, 0, stream>>>(src, dst, el, er, mbuf, s);
    k_edge_agg<<<NEDGES / 8, 256, 0, stream>>>(src, dst, el, er, mbuf, s, H, agg);
    k_head<<<(NNODES + 7) / 8, 256, 0, stream>>>(agg, bias, fc_w, fc_b, out);
}
